// Transformer_4260607557882
// MI455X (gfx1250) — matrix operand order repaired and checked by reading, not compiled, not run on hardware
//
#include <hip/hip_runtime.h>
#include <math.h>

#define DIMC 256
#define DEPTH 4
#define HEADS 8
#define DIM_HEAD 32
#define MLPD 1024
#define BB 16
#define NN 4096
#define EPSV 1e-5f

typedef __attribute__((ext_vector_type(16))) __bf16 v16bf;
typedef __attribute__((ext_vector_type(8)))  float  v8f;

union Frag { unsigned int u[8]; v16bf v; };

__device__ __forceinline__ unsigned short f2bf(float f) {
    unsigned int u = __float_as_uint(f);
    u += 0x7FFFu + ((u >> 16) & 1u);   // round-to-nearest-even
    return (unsigned short)(u >> 16);
}

// ---------------------------------------------------------------------------
// Pack W1/W2 (all layers) fp32 -> bf16 pairs in the exact WMMA B-fragment
// layout: dword p = (((kstep*NT + ntile)*32 + lane)*8 + v) holds bf16 pair
// (k0, k0+1) for column n = ntile*16 + (lane&15), k0 = kstep*32+((v&4)<<2)+((lane&16)>>1)+2(v&3).
// ---------------------------------------------------------------------------
__global__ void pack_w(const float* __restrict__ W1, const float* __restrict__ W2,
                       unsigned int* __restrict__ pW1, unsigned int* __restrict__ pW2) {
    const int PER = 131072;                     // dwords per layer per matrix
    int g = blockIdx.x * blockDim.x + threadIdx.x;
    int which = g / (DEPTH * PER);
    int rem   = g % (DEPTH * PER);
    int l = rem / PER;
    int p = rem % PER;
    int v = p & 7, lane = (p >> 3) & 31;
    if (which == 0) {                           // W1: [256 x 1024], 8 ksteps, 64 ntiles
        int ntile = (p >> 8) & 63, kstep = p >> 14;
        int k0 = kstep * 32 + ((v & 4) << 2) + ((lane & 16) >> 1) + 2 * (v & 3);
        int n  = ntile * 16 + (lane & 15);
        const float* src = W1 + (size_t)l * DIMC * MLPD;
        unsigned int lo = f2bf(src[(size_t)k0 * MLPD + n]);
        unsigned int hi = f2bf(src[(size_t)(k0 + 1) * MLPD + n]);
        pW1[(size_t)l * PER + p] = lo | (hi << 16);
    } else {                                    // W2: [1024 x 256], 32 ksteps, 16 ntiles
        int ntile = (p >> 8) & 15, kstep = p >> 12;
        int k0 = kstep * 32 + ((v & 4) << 2) + ((lane & 16) >> 1) + 2 * (v & 3);
        int n  = ntile * 16 + (lane & 15);
        const float* src = W2 + (size_t)l * MLPD * DIMC;
        unsigned int lo = f2bf(src[(size_t)k0 * DIMC + n]);
        unsigned int hi = f2bf(src[(size_t)(k0 + 1) * DIMC + n]);
        pW2[(size_t)l * PER + p] = lo | (hi << 16);
    }
}

// ---------------------------------------------------------------------------
// Per layer: gather center token, q = sel@Wq, qhat[h,c] = scale * Wk[c,h*32+d]·q[h*32+d]
// Also zeroes ctx for this layer.
// ---------------------------------------------------------------------------
__global__ void qhat_kernel(const float* __restrict__ xbuf, const int* __restrict__ pidx,
                            const int* __restrict__ scale, const float* __restrict__ Wq,
                            const float* __restrict__ Wk, float* __restrict__ qhat,
                            float* __restrict__ ctx) {
    int b = blockIdx.x, t = threadIdx.x;
    __shared__ float sel[DIMC], qv[DIMC];
    int idx = pidx[b * 2] * scale[1] + pidx[b * 2 + 1];
    sel[t] = xbuf[((size_t)b * NN + idx) * DIMC + t];
    for (int i = 0; i < 8; i++) ctx[b * HEADS * DIMC + i * 256 + t] = 0.f;
    __syncthreads();
    float acc = 0.f;
    for (int c = 0; c < DIMC; c++) acc += sel[c] * Wq[c * 256 + t];
    qv[t] = acc;
    __syncthreads();
    const float s = 0.17677669529663689f;       // 32^-0.5 folded into qhat
    for (int h = 0; h < HEADS; h++) {
        float a = 0.f;
        for (int d = 0; d < DIM_HEAD; d++)
            a += Wk[t * 256 + h * DIM_HEAD + d] * qv[h * DIM_HEAD + d];
        qhat[(b * HEADS + h) * DIMC + t] = a * s;
    }
}

// dots[b,h,n] = x[b,n,:] . qhat[b,h,:]   (memory-bound sweep over x)
__global__ void dots_kernel(const float* __restrict__ xbuf, const float* __restrict__ qhat,
                            float* __restrict__ dots) {
    int b = blockIdx.x, chunk = blockIdx.y, t = threadIdx.x;
    int n0 = chunk * 32;
    __shared__ float xs[32][DIMC + 1];
    __shared__ float qh[HEADS][DIMC + 1];
    for (int i = 0; i < 8; i++) {
        int li = t + i * 256;
        qh[li >> 8][li & 255] = qhat[b * HEADS * DIMC + li];
    }
    for (int i = 0; i < 32; i++)
        xs[i][t] = xbuf[((size_t)b * NN + n0 + i) * DIMC + t];
    __syncthreads();
    int tok = t >> 3, h = t & 7;
    float acc = 0.f;
    for (int c = 0; c < DIMC; c++) acc += xs[tok][c] * qh[h][c];
    dots[((size_t)b * HEADS + h) * NN + n0 + tok] = acc;
}

__global__ void softmax_kernel(float* __restrict__ dots) {
    int g = blockIdx.x, t = threadIdx.x;
    float* row = dots + (size_t)g * NN;
    __shared__ float red[256];
    float mx = -1e30f;
    for (int i = t; i < NN; i += 256) mx = fmaxf(mx, row[i]);
    red[t] = mx; __syncthreads();
    for (int s = 128; s > 0; s >>= 1) { if (t < s) red[t] = fmaxf(red[t], red[t + s]); __syncthreads(); }
    mx = red[0]; __syncthreads();
    float sum = 0.f;
    for (int i = t; i < NN; i += 256) { float e = expf(row[i] - mx); row[i] = e; sum += e; }
    red[t] = sum; __syncthreads();
    for (int s = 128; s > 0; s >>= 1) { if (t < s) red[t] += red[t + s]; __syncthreads(); }
    float inv = 1.f / red[0];
    __syncthreads();
    for (int i = t; i < NN; i += 256) row[i] *= inv;
}

// ctx[b,h,c] = sum_n attn[b,h,n] * x[b,n,c]  (second memory-bound sweep over x)
__global__ void ctx_kernel(const float* __restrict__ xbuf, const float* __restrict__ attn,
                           float* __restrict__ ctx) {
    int b = blockIdx.x, chunk = blockIdx.y, t = threadIdx.x;
    int n0 = chunk * 128;
    __shared__ float at[HEADS][128];
    for (int i = 0; i < 4; i++) {
        int li = t + i * 256; int h = li >> 7, n = li & 127;
        at[h][n] = attn[((size_t)b * HEADS + h) * NN + n0 + n];
    }
    __syncthreads();
    float acc[HEADS];
    for (int h = 0; h < HEADS; h++) acc[h] = 0.f;
    for (int n = 0; n < 128; n++) {
        float xv = xbuf[((size_t)b * NN + n0 + n) * DIMC + t];
        for (int h = 0; h < HEADS; h++) acc[h] += at[h][n] * xv;
    }
    for (int h = 0; h < HEADS; h++)
        atomicAdd(&ctx[(b * HEADS + h) * DIMC + t], acc[h]);
}

// o = ctx @ Wv (per head), final = o @ Wo + bo, x[b, idx] += final
__global__ void attn_out_scatter(float* __restrict__ xbuf, const float* __restrict__ ctx,
                                 const float* __restrict__ Wv, const float* __restrict__ Wo,
                                 const float* __restrict__ bo, const int* __restrict__ pidx,
                                 const int* __restrict__ scale) {
    int b = blockIdx.x, t = threadIdx.x;
    __shared__ float cs[HEADS * DIMC];
    __shared__ float o[256];
    for (int i = 0; i < 8; i++) cs[t + i * 256] = ctx[b * HEADS * DIMC + t + i * 256];
    __syncthreads();
    int h = t >> 5;
    float a = 0.f;
    for (int c = 0; c < DIMC; c++) a += cs[h * DIMC + c] * Wv[c * 256 + t];
    o[t] = a; __syncthreads();
    float f = bo[t];
    for (int i = 0; i < 256; i++) f += o[i] * Wo[i * DIMC + t];
    int idx = pidx[b * 2] * scale[1] + pidx[b * 2 + 1];
    xbuf[((size_t)b * NN + idx) * DIMC + t] += f;
}

// ---------------------------------------------------------------------------
// Fused MLP: x <- gelu(LN(x)@W1 + b1)@W2 + b2 + x.
// MT=32 tokens/block (2 M-tiles), 8 waves. Each wave owns a 128-col strip of H
// for BOTH M-tiles (B fragments reused 2x). GEMM1 runs in two N-passes of
// 4 N-tiles each so peak live accumulators stay at 64 VGPRs (no spills), with
// each GELU epilogue executed while pressure is low. B traffic is unchanged:
// every W1/W2 fragment is still read exactly once per wave.
// ---------------------------------------------------------------------------
#define MT   32
#define ASTR 264
#define HSTR 1032

__global__ void __launch_bounds__(256) mlp_kernel(
        float* __restrict__ xbuf, const unsigned int* __restrict__ pW1,
        const unsigned int* __restrict__ pW2, const float* __restrict__ b1,
        const float* __restrict__ b2, const float* __restrict__ lng,
        const float* __restrict__ lnb) {
    __shared__ float xin[MT * DIMC];            // 32 KB fp32 tile (kept for residual)
    __shared__ unsigned short aln[MT * ASTR];   // ~17 KB LN'd bf16 tile (GEMM1 A)
    __shared__ unsigned short Hs[MT * HSTR];    // ~66 KB bf16 hidden tile (GEMM2 A)
    __shared__ float redS[256], redQ[256];
    __shared__ float mu[MT], rs[MT];
    int t = threadIdx.x;
    int lane = t & 31, w = t >> 5;              // wave32
    size_t r0 = (size_t)blockIdx.x * MT;

    for (int i = 0; i < MT; i++) xin[t + i * 256] = xbuf[r0 * DIMC + t + i * 256];
    __syncthreads();

    // LayerNorm: 8 threads per row, 32 cols each
    {
        int row = t >> 3, c0 = (t & 7) * 32;
        float s = 0.f, q = 0.f;
        for (int j = 0; j < 32; j++) { float v = xin[row * DIMC + c0 + j]; s += v; q += v * v; }
        redS[t] = s; redQ[t] = q;
    }
    __syncthreads();
    if (t < MT) {
        float s = 0.f, q = 0.f;
        for (int i = 0; i < 8; i++) { s += redS[t * 8 + i]; q += redQ[t * 8 + i]; }
        float m = s * (1.f / DIMC);
        mu[t] = m;
        rs[t] = rsqrtf(q * (1.f / DIMC) - m * m + EPSV);
    }
    __syncthreads();
    {
        int row = t >> 3, c0 = (t & 7) * 32;
        float m = mu[row], r = rs[row];
        for (int j = 0; j < 32; j++) {
            int c = c0 + j;
            aln[row * ASTR + c] = f2bf((xin[row * DIMC + c] - m) * r * lng[c] + lnb[c]);
        }
    }
    __syncthreads();

    int m = lane & 15;
    int kb = (lane & 16) >> 1;                  // 0 or 8 (A-fragment K base)
    int mbase = (lane & 16) ? 8 : 0;            // C/D-fragment row base

    // ---- GEMM1: H = LN(x) @ W1, wave w -> cols [w*128, w*128+128) ----
    // Two passes of 4 N-tiles x 2 M-tiles (64 acc VGPRs live per pass).
    for (int half = 0; half < 2; half++) {
        v8f acc0[4] = {};
        v8f acc1[4] = {};
        for (int ks = 0; ks < 8; ks++) {
            Frag a0, a1;
            for (int vi = 0; vi < 8; vi++) {
                int k0 = ks * 32 + ((vi & 4) << 2) + kb + 2 * (vi & 3);
                a0.u[vi] = *(const unsigned int*)&aln[m * ASTR + k0];
                a1.u[vi] = *(const unsigned int*)&aln[(16 + m) * ASTR + k0];
            }
            for (int nt = 0; nt < 4; nt++) {
                int ntile = w * 8 + half * 4 + nt;
                Frag bf;
                const unsigned int* bp = pW1 + (((size_t)(ks * 64 + ntile) * 32 + lane) * 8);
                for (int vi = 0; vi < 8; vi++) bf.u[vi] = bp[vi];
                acc0[nt] = __builtin_amdgcn_wmma_f32_16x16x32_bf16(
                    false, a0.v, false, bf.v, (short)0, acc0[nt], false, false);
                acc1[nt] = __builtin_amdgcn_wmma_f32_16x16x32_bf16(
                    false, a1.v, false, bf.v, (short)0, acc1[nt], false, false);
            }
        }
        // bias + exact GELU for this 64-col strip, write bf16 H tile to LDS
        for (int nt = 0; nt < 4; nt++) {
            int n = w * 128 + (half * 4 + nt) * 16 + (lane & 15);
            float bv = b1[n];
            for (int r = 0; r < 8; r++) {
                float h0 = acc0[nt][r] + bv;
                float h1 = acc1[nt][r] + bv;
                float g0 = 0.5f * h0 * (1.f + erff(h0 * 0.70710678118654752f));
                float g1 = 0.5f * h1 * (1.f + erff(h1 * 0.70710678118654752f));
                Hs[(mbase + r) * HSTR + n]      = f2bf(g0);
                Hs[(16 + mbase + r) * HSTR + n] = f2bf(g1);
            }
        }
    }
    __syncthreads();

    // ---- GEMM2: OUT = H @ W2, wave w -> cols [w*32, w*32+32), 2 M-tiles ----
    v8f acc2[2][2] = {};
    for (int ks = 0; ks < 32; ks++) {
        Frag a0, a1;
        for (int vi = 0; vi < 8; vi++) {
            int k0 = ks * 32 + ((vi & 4) << 2) + kb + 2 * (vi & 3);
            a0.u[vi] = *(const unsigned int*)&Hs[m * HSTR + k0];
            a1.u[vi] = *(const unsigned int*)&Hs[(16 + m) * HSTR + k0];
        }
        for (int j = 0; j < 2; j++) {
            Frag bf;
            const unsigned int* bp = pW2 + (((size_t)(ks * 16 + w * 2 + j) * 32 + lane) * 8);
            for (int vi = 0; vi < 8; vi++) bf.u[vi] = bp[vi];
            acc2[0][j] = __builtin_amdgcn_wmma_f32_16x16x32_bf16(
                false, a0.v, false, bf.v, (short)0, acc2[0][j], false, false);
            acc2[1][j] = __builtin_amdgcn_wmma_f32_16x16x32_bf16(
                false, a1.v, false, bf.v, (short)0, acc2[1][j], false, false);
        }
    }
    for (int j = 0; j < 2; j++) {
        int n = (w * 2 + j) * 16 + (lane & 15);
        float bv = b2[n];
        for (int r = 0; r < 8; r++) {
            int mm0 = mbase + r;
            int mm1 = 16 + mbase + r;
            xbuf[(r0 + mm0) * DIMC + n] = acc2[0][j][r] + bv + xin[mm0 * DIMC + n];
            xbuf[(r0 + mm1) * DIMC + n] = acc2[1][j][r] + bv + xin[mm1 * DIMC + n];
        }
    }
}

__global__ void final_ln(const float* __restrict__ xbuf, const float* __restrict__ g,
                         const float* __restrict__ bb, float* __restrict__ out) {
    __shared__ float xin[16 * DIMC];
    __shared__ float redS[256], redQ[256];
    __shared__ float mu[16], rs[16];
    int t = threadIdx.x;
    size_t r0 = (size_t)blockIdx.x * 16;
    for (int i = 0; i < 16; i++) xin[t + i * 256] = xbuf[r0 * DIMC + t + i * 256];
    __syncthreads();
    int row = t >> 4, c0 = (t & 15) * 16;
    float s = 0.f, q = 0.f;
    for (int j = 0; j < 16; j++) { float v = xin[row * DIMC + c0 + j]; s += v; q += v * v; }
    redS[t] = s; redQ[t] = q; __syncthreads();
    if (t < 16) {
        float ss = 0.f, qq = 0.f;
        for (int i = 0; i < 16; i++) { ss += redS[t * 16 + i]; qq += redQ[t * 16 + i]; }
        float mm = ss * (1.f / DIMC);
        mu[t] = mm; rs[t] = rsqrtf(qq * (1.f / DIMC) - mm * mm + EPSV);
    }
    __syncthreads();
    float mm = mu[row], rr = rs[row];
    for (int j = 0; j < 16; j++) {
        int c = c0 + j;
        out[(r0 + row) * DIMC + c] = (xin[row * DIMC + c] - mm) * rr * g[c] + bb[c];
    }
}

// ---------------------------------------------------------------------------
extern "C" void kernel_launch(void* const* d_in, const int* in_sizes, int n_in,
                              void* d_out, int out_size, void* d_ws, size_t ws_size,
                              hipStream_t stream) {
    const float* x    = (const float*)d_in[0];
    const int*   pidx = (const int*)d_in[1];
    const int*   scl  = (const int*)d_in[3];
    const float* Wq   = (const float*)d_in[4];
    const float* Wk   = (const float*)d_in[5];
    const float* Wv   = (const float*)d_in[6];
    const float* Wo   = (const float*)d_in[7];
    const float* bo   = (const float*)d_in[8];
    const float* lng  = (const float*)d_in[9];
    const float* lnb  = (const float*)d_in[10];
    const float* W1   = (const float*)d_in[11];
    const float* b1   = (const float*)d_in[12];
    const float* W2   = (const float*)d_in[13];
    const float* b2   = (const float*)d_in[14];
    const float* lfg  = (const float*)d_in[15];
    const float* lfb  = (const float*)d_in[16];

    char* ws = (char*)d_ws;
    float*        xbuf = (float*)ws;                          // 67,108,864 B
    float*        dots = (float*)(ws + 67108864);             //  2,097,152 B
    float*        qhat = (float*)(ws + 69206016);             //    131,072 B
    float*        ctx  = (float*)(ws + 69337088);             //    131,072 B
    unsigned int* pW1  = (unsigned int*)(ws + 69468160);      //  2,097,152 B
    unsigned int* pW2  = (unsigned int*)(ws + 71565312);      //  2,097,152 B

    hipMemcpyAsync(xbuf, x, (size_t)BB * NN * DIMC * sizeof(float),
                   hipMemcpyDeviceToDevice, stream);
    pack_w<<<4096, 256, 0, stream>>>(W1, W2, pW1, pW2);

    for (int l = 0; l < DEPTH; l++) {
        qhat_kernel<<<BB, 256, 0, stream>>>(xbuf, pidx, scl,
            Wq + (size_t)l * DIMC * 256, Wk + (size_t)l * DIMC * 256, qhat, ctx);
        dots_kernel<<<dim3(BB, NN / 32), 256, 0, stream>>>(xbuf, qhat, dots);
        softmax_kernel<<<BB * HEADS, 256, 0, stream>>>(dots);
        ctx_kernel<<<dim3(BB, NN / 128), 256, 0, stream>>>(xbuf, dots, ctx);
        attn_out_scatter<<<BB, 256, 0, stream>>>(xbuf, ctx,
            Wv + (size_t)l * DIMC * 256, Wo + (size_t)l * 256 * DIMC,
            bo + l * DIMC, pidx, scl);
        mlp_kernel<<<(BB * NN) / MT, 256, 0, stream>>>(xbuf,
            pW1 + (size_t)l * 131072, pW2 + (size_t)l * 131072,
            b1 + l * MLPD, b2 + l * DIMC, lng + l * DIMC, lnb + l * DIMC);
    }
    final_ln<<<(BB * NN) / 16, 256, 0, stream>>>(xbuf, lfg, lfb, (float*)d_out);
}
